// Node_Edge_cls_Module_75788992905487
// MI455X (gfx1250) — compile-verified, matrix operand order checked by reading, not run on hardware
//
#include <hip/hip_runtime.h>
#include <hip/hip_bf16.h>

// ---------------------------------------------------------------------------
// Problem constants (match reference)
// ---------------------------------------------------------------------------
#define INC 128
#define HID 256
#define OUTC 256
#define NODE_CLS 7
#define EDGE_CLS 2
#define SG_CLS 5
#define SLOPE 0.2f

typedef __attribute__((ext_vector_type(16))) __bf16 v16bf;
typedef __attribute__((ext_vector_type(8)))  __bf16 v8bf;   // 16 bytes = one b128
typedef __attribute__((ext_vector_type(8)))  float  v8f;

__device__ __forceinline__ float lrelu(float x) { return x > 0.f ? x : SLOPE * x; }
__device__ __forceinline__ float eluf(float x)  { return x > 0.f ? x : (__expf(x) - 1.f); }

// order-preserving float<->uint encoding for atomic max on floats
__device__ __forceinline__ unsigned fenc(float f) {
    unsigned u = __float_as_uint(f);
    return (u & 0x80000000u) ? ~u : (u | 0x80000000u);
}
__device__ __forceinline__ float fdec(unsigned u) {
    unsigned v = (u & 0x80000000u) ? (u & 0x7FFFFFFFu) : ~u;
    return __uint_as_float(v);
}

enum { AMODE_PLAIN = 0, AMODE_GATHER2 = 1, AMODE_GATHER3 = 2 };
enum { ACTA_NONE = 0, ACTA_ELU = 1 };
enum { EPI_BIAS = 0, EPI_RELU = 1, EPI_GATE = 2, EPI_ADD2_LRELU = 3 };

__device__ __forceinline__ int kmap(int v, int h) {
    // 16-bit A/B fragment K index per CDNA5 ISA 7.12.2 (16x32 layout)
    return ((v < 4) ? (2 * v) : (16 + 2 * (v - 4))) + 8 * h;
}

// ---------------------------------------------------------------------------
// Weight pack: W[K,N] f32 (row-major) -> fragment-ordered bf16 so each lane's
// 16-element B fragment is one contiguous 32-byte chunk.
// idx = ((kblk*(N/16)+cg)*32 + lane)*16 + 2*v + p
// ---------------------------------------------------------------------------
__global__ void k_pack_w(const float* __restrict__ W, __bf16* __restrict__ PW, int K, int N) {
    int idx = blockIdx.x * blockDim.x + threadIdx.x;
    if (idx >= K * N) return;
    int e    = idx & 15;
    int lane = (idx >> 4) & 31;
    int rest = idx >> 9;
    int NG   = N >> 4;
    int cg   = rest % NG;
    int kblk = rest / NG;
    int v = e >> 1, p = e & 1, h = lane >> 4;
    int c = cg * 16 + (lane & 15);
    int k = kblk * 32 + kmap(v, h) + p;
    PW[idx] = (__bf16)W[(size_t)k * N + c];
}

// ---------------------------------------------------------------------------
// WMMA GEMM:  C[M,Ncols] = epi( actA(A)[M,K] @ W[K,Ncols] + bias )
// wave32: wave owns a 32-row x 64-col tile (2 A frags, 4 B frags, 8 WMMA/kstep)
// A is f32 (b128 loads, bf16 convert in-register); PW is fragment-packed bf16.
// Out-of-range rows are CLAMPED at load (no exec predication in the hot loop)
// and discarded at store.
// ---------------------------------------------------------------------------
template <int AMODE, int ACTA, int EPI>
__global__ __launch_bounds__(128) void wmma_gemm(
    const float* __restrict__ A, int lda,
    const int* __restrict__ g0, const int* __restrict__ g1,
    const float* __restrict__ A2,
    const __bf16* __restrict__ PW, const float* __restrict__ bias,
    const float* __restrict__ gate, const float* __restrict__ nal,
    float* __restrict__ C, int M, int K, int Ncols)
{
    const int lane    = threadIdx.x & 31;
    const int wave    = threadIdx.x >> 5;
    const int rowbase = (blockIdx.x * 4 + wave) * 32;
    const int colbase = blockIdx.y * 64;
    const int half    = lane >> 4;
    const int l15     = lane & 15;
    const int NG      = Ncols >> 4;

    int aload[2]; int sidx[2], didx[2];
#pragma unroll
    for (int f = 0; f < 2; ++f) {
        aload[f] = min(rowbase + 16 * f + l15, M - 1);   // clamp: loads always valid
        sidx[f] = 0; didx[f] = 0;
        if (AMODE != AMODE_PLAIN) { sidx[f] = g0[aload[f]]; didx[f] = g1[aload[f]]; }
    }

    v8f acc[2][4];
#pragma unroll
    for (int f = 0; f < 2; ++f)
#pragma unroll
        for (int j = 0; j < 4; ++j) acc[f][j] = (v8f){};

    const v8bf* pw = (const v8bf*)PW;

    for (int kb = 0; kb < K; kb += 32) {
        // ----- A fragments: two contiguous 8-float runs per lane -----
        v16bf af[2];
#pragma unroll
        for (int f = 0; f < 2; ++f) {
            const float* rp;
            if (AMODE == AMODE_PLAIN) {
                rp = A + (size_t)aload[f] * lda + kb;
            } else {
                int kk = kb; const float* base = A; int r = sidx[f];
                if (AMODE == AMODE_GATHER3 && kk >= 512) { base = A2; r = aload[f]; kk -= 512; }
                else if (kk >= 256)                       { r = didx[f]; kk -= 256; }
                rp = base + (size_t)r * 256 + kk;
            }
            float4 u0 = *(const float4*)(rp + 8 * half);
            float4 u1 = *(const float4*)(rp + 8 * half + 4);
            float4 u2 = *(const float4*)(rp + 8 * half + 16);
            float4 u3 = *(const float4*)(rp + 8 * half + 20);
            if (AMODE == AMODE_PLAIN && kb + 32 < K) __builtin_prefetch(rp + 32, 0, 3);
#define CVT1(x) ((__bf16)((ACTA == ACTA_ELU) ? eluf(x) : (x)))
            af[f][0]  = CVT1(u0.x); af[f][1]  = CVT1(u0.y);
            af[f][2]  = CVT1(u0.z); af[f][3]  = CVT1(u0.w);
            af[f][4]  = CVT1(u1.x); af[f][5]  = CVT1(u1.y);
            af[f][6]  = CVT1(u1.z); af[f][7]  = CVT1(u1.w);
            af[f][8]  = CVT1(u2.x); af[f][9]  = CVT1(u2.y);
            af[f][10] = CVT1(u2.z); af[f][11] = CVT1(u2.w);
            af[f][12] = CVT1(u3.x); af[f][13] = CVT1(u3.y);
            af[f][14] = CVT1(u3.z); af[f][15] = CVT1(u3.w);
#undef CVT1
        }
        // ----- B fragments: fragment-packed, two b128 loads each -----
        const size_t pb = ((size_t)(kb >> 5) * NG + (colbase >> 4)) * 64 + (size_t)lane * 2;
        v16bf bf[4];
#pragma unroll
        for (int j = 0; j < 4; ++j) {
            v8bf lo = pw[pb + (size_t)j * 64];
            v8bf hi = pw[pb + (size_t)j * 64 + 1];
            bf[j] = __builtin_shufflevector(lo, hi, 0, 1, 2, 3, 4, 5, 6, 7,
                                            8, 9, 10, 11, 12, 13, 14, 15);
        }
#pragma unroll
        for (int f = 0; f < 2; ++f)
#pragma unroll
            for (int j = 0; j < 4; ++j)
                acc[f][j] = __builtin_amdgcn_wmma_f32_16x16x32_bf16(
                    false, af[f], false, bf[j], (short)0, acc[f][j], false, false);
    }

    // ----- store: frag f, VGPR i -> row = rowbase+16f+8*half+i, col = colbase+16j+l15
#pragma unroll
    for (int f = 0; f < 2; ++f) {
        const int crow0 = rowbase + 16 * f + 8 * half;
#pragma unroll
        for (int i = 0; i < 8; ++i) {
            const int r = crow0 + i;
            if (r >= M) continue;
            float g = 1.f;
            int s2 = 0, d2 = 0;
            if (EPI == EPI_GATE) g = gate[r];
            if (EPI == EPI_ADD2_LRELU) { s2 = g0[r]; d2 = g1[r]; }
#pragma unroll
            for (int j = 0; j < 4; ++j) {
                const int col = colbase + 16 * j + l15;
                float v = acc[f][j][i];
                if (bias) v += bias[col];
                if (EPI == EPI_RELU)       v = v > 0.f ? v : 0.f;
                if (EPI == EPI_GATE)       v *= g;
                if (EPI == EPI_ADD2_LRELU) v = lrelu(v + nal[(size_t)s2 * 256 + col] + nal[(size_t)d2 * 256 + col]);
                C[(size_t)r * Ncols + col] = v;
            }
        }
    }
}

// ---------------------------------------------------------------------------
// Helper kernels
// ---------------------------------------------------------------------------
__global__ void k_zero_u32(unsigned* p, int n) {
    int i = blockIdx.x * blockDim.x + threadIdx.x;
    if (i < n) p[i] = 0u;
}
__global__ void k_zero_f32(float* p, int n) {
    int i = blockIdx.x * blockDim.x + threadIdx.x;
    if (i < n) p[i] = 0.f;
}
__global__ void k_fill_rows256(float* __restrict__ dst, const float* __restrict__ bvec, int total) {
    int i = blockIdx.x * blockDim.x + threadIdx.x;
    if (i < total) dst[i] = bvec[i & 255];
}
// wave-per-node coalesced dual dot product
__global__ void k_att_dots(const float* __restrict__ h, const float* __restrict__ av,
                           const float* __restrict__ bv, float* __restrict__ a_s,
                           float* __restrict__ a_d, int n) {
    int lane = threadIdx.x & 31;
    int node = (blockIdx.x * blockDim.x + threadIdx.x) >> 5;
    if (node >= n) return;
    const float* hp = h + (size_t)node * HID;
    float s = 0.f, d = 0.f;
#pragma unroll
    for (int t = 0; t < HID / 32; ++t) {
        int k = lane + 32 * t;
        float v = hp[k];
        s += v * av[k]; d += v * bv[k];
    }
#pragma unroll
    for (int off = 16; off; off >>= 1) { s += __shfl_xor(s, off, 32); d += __shfl_xor(d, off, 32); }
    if (lane == 0) { a_s[node] = s; a_d[node] = d; }
}
__global__ void k_edge_max(const int* __restrict__ src, const int* __restrict__ dst,
                           const float* __restrict__ a_s, const float* __restrict__ a_d,
                           float* __restrict__ ew, unsigned* __restrict__ menc, int E, int N) {
    int i = blockIdx.x * blockDim.x + threadIdx.x;
    int T = E + N;
    if (i >= T) return;
    int s = i < E ? src[i] : i - E;
    int d = i < E ? dst[i] : i - E;
    float e = lrelu(a_s[s] + a_d[d]);
    ew[i] = e;
    atomicMax(&menc[d], fenc(e));
}
__global__ void k_edge_w(const int* __restrict__ dst, const unsigned* __restrict__ menc,
                         float* __restrict__ ew, float* __restrict__ denom, int E, int N) {
    int i = blockIdx.x * blockDim.x + threadIdx.x;
    int T = E + N;
    if (i >= T) return;
    int d = i < E ? dst[i] : i - E;
    float w = __expf(ew[i] - fdec(menc[d]));
    ew[i] = w;
    atomicAdd(&denom[d], w);
}
__global__ void k_scatter(const int* __restrict__ src, const int* __restrict__ dst,
                          const float* __restrict__ ew, const float* __restrict__ denom,
                          const float* __restrict__ h, float* __restrict__ gat, int E, int N) {
    int lane = threadIdx.x & 31;
    int wid  = (blockIdx.x * blockDim.x + threadIdx.x) >> 5;
    int T = E + N;
    if (wid >= T) return;
    int s = wid < E ? src[wid] : wid - E;
    int d = wid < E ? dst[wid] : wid - E;
    float alpha = ew[wid] / (denom[d] + 1e-16f);
    const float* hs = h + (size_t)s * HID;
    float* gd = gat + (size_t)d * HID;
#pragma unroll
    for (int t = 0; t < 8; ++t) {
        int c = lane + 32 * t;
        atomicAdd(&gd[c], alpha * hs[c]);
    }
}
// wave-per-row small classifier head (Ncls <= 8), coalesced
__global__ void k_small_out(const float* __restrict__ in, const float* __restrict__ W,
                            const float* __restrict__ b, float* __restrict__ out,
                            int rows, int K, int Ncls) {
    int lane = threadIdx.x & 31;
    int r    = (blockIdx.x * blockDim.x + threadIdx.x) >> 5;
    if (r >= rows) return;
    const float* x = in + (size_t)r * K;
    float acc[8];
#pragma unroll
    for (int c = 0; c < 8; ++c) acc[c] = 0.f;
    for (int k = lane; k < K; k += 32) {
        float xv = x[k];
#pragma unroll
        for (int c = 0; c < 8; ++c)
            if (c < Ncls) acc[c] += xv * W[k * Ncls + c];
    }
#pragma unroll
    for (int c = 0; c < 8; ++c) {
        if (c >= Ncls) break;
        float s = acc[c];
#pragma unroll
        for (int off = 16; off; off >>= 1) s += __shfl_xor(s, off, 32);
        if (lane == 0) out[(size_t)r * Ncls + c] = s + b[c];
    }
}
__global__ void k_gate(const int* __restrict__ src, const int* __restrict__ dst,
                       const float* __restrict__ ne, const float* __restrict__ ee,
                       const float* __restrict__ Wa, const float* __restrict__ ba,
                       float* __restrict__ gate, int E) {
    int lane = threadIdx.x & 31;
    int e    = (blockIdx.x * blockDim.x + threadIdx.x) >> 5;
    if (e >= E) return;
    int s = src[e], d = dst[e];
    float sum = 0.f;
#pragma unroll
    for (int t = 0; t < 24; ++t) {
        int k = lane + 32 * t;
        float v = (k < 256) ? ne[(size_t)s * 256 + k]
                : (k < 512) ? ne[(size_t)d * 256 + (k - 256)]
                            : ee[(size_t)e * 256 + (k - 512)];
        sum += v * Wa[k];
    }
#pragma unroll
    for (int off = 16; off; off >>= 1) sum += __shfl_xor(sum, off, 32);
    if (lane == 0) {
        float z = lrelu(sum + ba[0]);
        gate[e] = 1.f / (1.f + __expf(-z));
    }
}
// column softmax over edges, channel = threadIdx.x (256 channels, coalesced)
__global__ void k_colmax(const float* __restrict__ S, unsigned* __restrict__ cmax, int E, int rpb) {
    int c = threadIdx.x;
    int r0 = blockIdx.x * rpb, r1 = min(E, r0 + rpb);
    float m = -3.4e38f;
    for (int r = r0; r < r1; ++r) m = fmaxf(m, S[(size_t)r * 256 + c]);
    atomicMax(&cmax[c], fenc(m));
}
__global__ void k_colsum(const float* __restrict__ S, const unsigned* __restrict__ cmax,
                         float* __restrict__ csum, int E, int rpb) {
    int c = threadIdx.x;
    int r0 = blockIdx.x * rpb, r1 = min(E, r0 + rpb);
    float mm = fdec(cmax[c]), s = 0.f;
    for (int r = r0; r < r1; ++r) s += __expf(S[(size_t)r * 256 + c] - mm);
    atomicAdd(&csum[c], s);
}
__global__ void k_refine(const float* __restrict__ S, const float* __restrict__ U,
                         const unsigned* __restrict__ cmax, const float* __restrict__ csum,
                         float* __restrict__ refined, float* __restrict__ pool, int E, int rpb) {
    int c = threadIdx.x;
    int r0 = blockIdx.x * rpb, r1 = min(E, r0 + rpb);
    float mm = fdec(cmax[c]), inv = 1.f / csum[c], acc = 0.f;
    for (int r = r0; r < r1; ++r) {
        float v = U[(size_t)r * 256 + c] * __expf(S[(size_t)r * 256 + c] - mm) * inv;
        refined[(size_t)r * 256 + c] = v;
        acc += v;
    }
    atomicAdd(&pool[c], acc);
}
__global__ void k_sg(const float* __restrict__ pool, const float* __restrict__ Wsg,
                     const float* __restrict__ bsg, float* __restrict__ out, int E) {
    int t = threadIdx.x;
    if (t >= SG_CLS) return;
    float s = bsg[t], inv = 1.f / (float)E;
    for (int c = 0; c < 256; ++c) s += pool[c] * inv * Wsg[c * SG_CLS + t];
    out[t] = s;
}

// ---------------------------------------------------------------------------
// launch
// ---------------------------------------------------------------------------
static inline int cdiv(int a, int b) { return (a + b - 1) / b; }

extern "C" void kernel_launch(void* const* d_in, const int* in_sizes, int n_in,
                              void* d_out, int out_size, void* d_ws, size_t ws_size,
                              hipStream_t stream) {
    const int N = in_sizes[0] / INC;       // 20000
    const int E = in_sizes[1] / 2;         // 200000

    const float* x        = (const float*)d_in[0];
    const int*   ei       = (const int*)d_in[1];
    const int*   src      = ei;
    const int*   dst      = ei + E;
    const float* W_gat    = (const float*)d_in[3];
    const float* att_src  = (const float*)d_in[4];
    const float* att_dst  = (const float*)d_in[5];
    const float* b_gat    = (const float*)d_in[6];
    const float* W_ne     = (const float*)d_in[7];
    const float* b_ne     = (const float*)d_in[8];
    const float* W_nc     = (const float*)d_in[9];
    const float* b_nc     = (const float*)d_in[10];
    const float* W_e1     = (const float*)d_in[11];
    const float* b_e1     = (const float*)d_in[12];
    const float* W_e2     = (const float*)d_in[13];
    const float* b_e2     = (const float*)d_in[14];
    const float* W_ec     = (const float*)d_in[15];
    const float* b_ec     = (const float*)d_in[16];
    const float* W_attn   = (const float*)d_in[17];
    const float* b_attn   = (const float*)d_in[18];
    const float* W_upd    = (const float*)d_in[19];
    const float* b_upd    = (const float*)d_in[20];
    const float* W_eal    = (const float*)d_in[21];
    const float* b_eal    = (const float*)d_in[22];
    const float* W_nal    = (const float*)d_in[23];
    const float* b_nal    = (const float*)d_in[24];
    const float* W_sg     = (const float*)d_in[25];
    const float* b_sg     = (const float*)d_in[26];

    float* out = (float*)d_out;
    float* out_node_logits = out;                                    // N*7
    float* out_edge_logits = out + (size_t)N * NODE_CLS;             // E*2
    float* out_refined     = out_edge_logits + (size_t)E * EDGE_CLS; // E*256
    float* out_sg          = out_refined + (size_t)E * 256;          // 5

    // ---- carve workspace ----
    char* p = (char*)d_ws;
    auto carve = [&](size_t bytes) -> void* {
        void* r = (void*)p;
        p += (bytes + 255) & ~(size_t)255;
        return r;
    };
    float*    h        = (float*)carve((size_t)N * HID * 4);
    float*    gat_out  = (float*)carve((size_t)N * HID * 4);
    float*    node_emb = (float*)carve((size_t)N * OUTC * 4);
    float*    node_nal = (float*)carve((size_t)N * 256 * 4);
    float*    a_s      = (float*)carve((size_t)N * 4);
    float*    a_d      = (float*)carve((size_t)N * 4);
    unsigned* menc     = (unsigned*)carve((size_t)N * 4);
    float*    denom    = (float*)carve((size_t)N * 4);
    float*    ew       = (float*)carve((size_t)(E + N) * 4);
    float*    t1       = (float*)carve((size_t)E * 256 * 4);   // reused as scores
    float*    edge_emb = (float*)carve((size_t)E * 256 * 4);
    float*    updated  = (float*)carve((size_t)E * 256 * 4);
    float*    gate     = (float*)carve((size_t)E * 4);
    unsigned* cmax     = (unsigned*)carve(256 * 4);
    float*    csum     = (float*)carve(256 * 4);
    float*    pool     = (float*)carve(256 * 4);
    __bf16* Wb_gat = (__bf16*)carve((size_t)INC * HID * 2);
    __bf16* Wb_ne  = (__bf16*)carve((size_t)HID * OUTC * 2);
    __bf16* Wb_e1  = (__bf16*)carve((size_t)512 * 256 * 2);
    __bf16* Wb_e2  = (__bf16*)carve((size_t)256 * 256 * 2);
    __bf16* Wb_upd = (__bf16*)carve((size_t)768 * 256 * 2);
    __bf16* Wb_eal = (__bf16*)carve((size_t)256 * 256 * 2);
    __bf16* Wb_nal = (__bf16*)carve((size_t)256 * 256 * 2);
    (void)ws_size; (void)n_in; (void)out_size;

    // ---- 0. weights -> fragment-packed bf16 ----
    k_pack_w<<<cdiv(INC * HID, 256), 256, 0, stream>>>(W_gat, Wb_gat, INC, HID);
    k_pack_w<<<cdiv(HID * OUTC, 256), 256, 0, stream>>>(W_ne, Wb_ne, HID, OUTC);
    k_pack_w<<<cdiv(512 * 256, 256), 256, 0, stream>>>(W_e1, Wb_e1, 512, 256);
    k_pack_w<<<cdiv(256 * 256, 256), 256, 0, stream>>>(W_e2, Wb_e2, 256, 256);
    k_pack_w<<<cdiv(768 * 256, 256), 256, 0, stream>>>(W_upd, Wb_upd, 768, 256);
    k_pack_w<<<cdiv(256 * 256, 256), 256, 0, stream>>>(W_eal, Wb_eal, 256, 256);
    k_pack_w<<<cdiv(256 * 256, 256), 256, 0, stream>>>(W_nal, Wb_nal, 256, 256);

    // ---- 1. h = x @ W_gat ----
    {
        dim3 g(cdiv(N, 128), HID / 64);
        wmma_gemm<AMODE_PLAIN, ACTA_NONE, EPI_BIAS><<<g, 128, 0, stream>>>(
            x, INC, nullptr, nullptr, nullptr, Wb_gat, nullptr, nullptr, nullptr,
            h, N, INC, HID);
    }
    // ---- 2. attention scalars + segment softmax + scatter ----
    k_att_dots<<<cdiv(N * 32, 256), 256, 0, stream>>>(h, att_src, att_dst, a_s, a_d, N);
    k_zero_u32<<<cdiv(N, 256), 256, 0, stream>>>(menc, N);
    k_zero_f32<<<cdiv(N, 256), 256, 0, stream>>>(denom, N);
    k_fill_rows256<<<cdiv(N * HID, 256), 256, 0, stream>>>(gat_out, b_gat, N * HID);
    k_edge_max<<<cdiv(E + N, 256), 256, 0, stream>>>(src, dst, a_s, a_d, ew, menc, E, N);
    k_edge_w<<<cdiv(E + N, 256), 256, 0, stream>>>(dst, menc, ew, denom, E, N);
    k_scatter<<<cdiv((E + N) * 32, 256), 256, 0, stream>>>(src, dst, ew, denom, h, gat_out, E, N);

    // ---- 3. node_emb = elu(gat_out) @ W_ne + b_ne ----
    {
        dim3 g(cdiv(N, 128), OUTC / 64);
        wmma_gemm<AMODE_PLAIN, ACTA_ELU, EPI_BIAS><<<g, 128, 0, stream>>>(
            gat_out, HID, nullptr, nullptr, nullptr, Wb_ne, b_ne, nullptr, nullptr,
            node_emb, N, HID, OUTC);
    }
    // ---- 4. node_logits ----
    k_small_out<<<cdiv(N * 32, 256), 256, 0, stream>>>(node_emb, W_nc, b_nc, out_node_logits, N, OUTC, NODE_CLS);

    // ---- 5. node_nal = node_emb @ W_nal + b_nal (per-node factorization) ----
    {
        dim3 g(cdiv(N, 128), 256 / 64);
        wmma_gemm<AMODE_PLAIN, ACTA_NONE, EPI_BIAS><<<g, 128, 0, stream>>>(
            node_emb, OUTC, nullptr, nullptr, nullptr, Wb_nal, b_nal, nullptr, nullptr,
            node_nal, N, OUTC, 256);
    }
    // ---- 6. t1 = relu([ne[src],ne[dst]] @ W_e1 + b_e1) ----
    {
        dim3 g(cdiv(E, 128), 256 / 64);
        wmma_gemm<AMODE_GATHER2, ACTA_NONE, EPI_RELU><<<g, 128, 0, stream>>>(
            node_emb, 0, src, dst, nullptr, Wb_e1, b_e1, nullptr, nullptr,
            t1, E, 512, 256);
    }
    // ---- 7. edge_emb = t1 @ W_e2 + b_e2 ----
    {
        dim3 g(cdiv(E, 128), 256 / 64);
        wmma_gemm<AMODE_PLAIN, ACTA_NONE, EPI_BIAS><<<g, 128, 0, stream>>>(
            t1, 256, nullptr, nullptr, nullptr, Wb_e2, b_e2, nullptr, nullptr,
            edge_emb, E, 256, 256);
    }
    // ---- 8. edge_logits ----
    k_small_out<<<cdiv(E * 32, 256), 256, 0, stream>>>(edge_emb, W_ec, b_ec, out_edge_logits, E, 256, EDGE_CLS);

    // ---- 9. gate = sigmoid(lrelu(comb3 @ W_attn + b_attn)) ----
    k_gate<<<cdiv(E * 32, 256), 256, 0, stream>>>(src, dst, node_emb, edge_emb, W_attn, b_attn, gate, E);

    // ---- 10. updated = (comb3 @ W_upd + b_upd) * gate ----
    {
        dim3 g(cdiv(E, 128), 256 / 64);
        wmma_gemm<AMODE_GATHER3, ACTA_NONE, EPI_GATE><<<g, 128, 0, stream>>>(
            node_emb, 0, src, dst, edge_emb, Wb_upd, b_upd, gate, nullptr,
            updated, E, 768, 256);
    }
    // ---- 11. scores = lrelu(updated @ W_eal + b_eal + nal[src] + nal[dst]) (t1 reused)
    {
        dim3 g(cdiv(E, 128), 256 / 64);
        wmma_gemm<AMODE_PLAIN, ACTA_NONE, EPI_ADD2_LRELU><<<g, 128, 0, stream>>>(
            updated, 256, src, dst, nullptr, Wb_eal, b_eal, nullptr, node_nal,
            t1, E, 256, 256);
    }
    // ---- 12. column softmax over edges + refined + pooled head ----
    k_zero_u32<<<1, 256, 0, stream>>>(cmax, 256);
    k_zero_f32<<<1, 256, 0, stream>>>(csum, 256);
    k_zero_f32<<<1, 256, 0, stream>>>(pool, 256);
    const int rpb = 256;
    k_colmax<<<cdiv(E, rpb), 256, 0, stream>>>(t1, cmax, E, rpb);
    k_colsum<<<cdiv(E, rpb), 256, 0, stream>>>(t1, cmax, csum, E, rpb);
    k_refine<<<cdiv(E, rpb), 256, 0, stream>>>(t1, updated, cmax, csum, out_refined, pool, E, rpb);
    k_sg<<<1, 32, 0, stream>>>(pool, W_sg, b_sg, out_sg, E);
}
